// JordanLayer_51977694216476
// MI455X (gfx1250) — compile-verified, matrix operand order checked by reading, not run on hardware
//
#include <hip/hip_runtime.h>
#include <hip/hip_bf16.h>
#include <math.h>

// Jordan RNN layer for MI455X (gfx1250), wave32 + bf16 WMMA.
// B=64, T=512, I=H=O=512.

typedef __bf16 bf16_t;
typedef bf16_t v16bf __attribute__((ext_vector_type(16)));
typedef bf16_t v8bf  __attribute__((ext_vector_type(8)));
typedef float  v8f   __attribute__((ext_vector_type(8)));

#define B_    64
#define T_    512
#define K_    512          // I == H == O == 512
#define LDSTR 520          // LDS row stride (elements): 1040B = 65*16B, bank-skewed

// Build a v16bf WMMA operand from two contiguous 16-byte chunks.
// Operand layout (A 16x32 / B 32x16 bf16): lane&15 selects the row (M for A,
// N for B); elements [0..7] are K = kk + 8*(lane>>4) .. +7, elements [8..15]
// are K = kk + 16 + 8*(lane>>4) .. +7.
__device__ __forceinline__ v16bf make16(const bf16_t* p0, const bf16_t* p1) {
  v8bf lo = *reinterpret_cast<const v8bf*>(p0);
  v8bf hi = *reinterpret_cast<const v8bf*>(p1);
  return __builtin_shufflevector(lo, hi, 0,1,2,3,4,5,6,7,8,9,10,11,12,13,14,15);
}

__device__ __forceinline__ v8f wmma_bf16(v16bf a, v16bf b, v8f c) {
  return __builtin_amdgcn_wmma_f32_16x16x32_bf16(
      /*neg_a=*/false, a, /*neg_b=*/false, b,
      /*c_mod=*/(short)0, c, /*reuse_a=*/false, /*reuse_b=*/false);
}

// ---------------------------------------------------------------------------
// Kernel 0: one-time f32 -> bf16 conversion of a weight matrix (RNE).
// ---------------------------------------------------------------------------
__global__ void cvt_bf16_kernel(const float* __restrict__ src,
                                bf16_t* __restrict__ dst, int n) {
  int i = blockIdx.x * blockDim.x + threadIdx.x;
  if (i < n) dst[i] = (bf16_t)src[i];
}

// ---------------------------------------------------------------------------
// Kernel 1: xin[bt, h] = sum_i emb[bt, i] * W_in[h, i] + b_in[h]
// Grid: 1024 blocks x 256 threads (8 waves). Each block computes a 32x512
// output strip (2 M-tiles); each wave owns 4 N-tiles and reuses every B
// operand across both M-tiles -> 8 WMMAs per 8 global b128 loads, and L2
// weight traffic is halved vs a 16-row tile.  A-strip staged in LDS as bf16.
// Result written f32 into the `h` region of d_out (consumed by the scan).
// ---------------------------------------------------------------------------
__global__ __launch_bounds__(256) void xin_gemm_kernel(
    const float* __restrict__ emb, const bf16_t* __restrict__ Wb,
    const float* __restrict__ bin, float* __restrict__ xout) {
  __shared__ bf16_t sA[32 * LDSTR];

  const int tid  = threadIdx.x;
  const int wave = tid >> 5;
  const int lane = tid & 31;
  const int half = lane >> 4;
  const int l15  = lane & 15;
  const size_t mbase = (size_t)blockIdx.x * 32;   // flat (b*T + t) row base

  for (int i = tid; i < 32 * K_; i += blockDim.x) {
    int r = i >> 9, c = i & (K_ - 1);
    sA[r * LDSTR + c] = (bf16_t)emb[(mbase + r) * K_ + c];
  }
  __syncthreads();

  const int n0 = wave * 64;
  v8f acc[2][4] = {};
#pragma unroll
  for (int kk = 0; kk < K_; kk += 32) {
    const bf16_t* ap0 = &sA[l15 * LDSTR + kk + 8 * half];
    const bf16_t* ap1 = ap0 + 16 * LDSTR;
    v16bf a0 = make16(ap0, ap0 + 16);
    v16bf a1 = make16(ap1, ap1 + 16);
#pragma unroll
    for (int tI = 0; tI < 4; ++tI) {
      const bf16_t* bp = &Wb[(size_t)(n0 + tI * 16 + l15) * K_ + kk + 8 * half];
      v16bf b = make16(bp, bp + 16);
      acc[0][tI] = wmma_bf16(a0, b, acc[0][tI]);
      acc[1][tI] = wmma_bf16(a1, b, acc[1][tI]);
    }
  }
#pragma unroll
  for (int mt = 0; mt < 2; ++mt) {
#pragma unroll
    for (int tI = 0; tI < 4; ++tI) {
      const int col = n0 + tI * 16 + l15;
      const float bias = bin[col];
#pragma unroll
      for (int r = 0; r < 8; ++r) {
        const int bl = mt * 16 + half * 8 + r;   // C layout: M = 8*half + r
        xout[(mbase + bl) * K_ + col] = acc[mt][tI][r] + bias;
      }
    }
  }
}

// ---------------------------------------------------------------------------
// Kernel 2: the sequential Jordan scan.  The recurrence is independent per
// batch row, so 4 persistent workgroups each own a 16-row batch slice and run
// all T steps locally (workgroup barriers only).  Recurrent state lives in
// LDS; W_y / W_out stream as bf16 from L2 (192 MB, trivially resident) every
// step.
//   phase 1: h = tanh(xin + y_prev @ W_y^T + b_y)   (h overwrites xin in d_out)
//   phase 2: y = h @ W_out^T + b_out
// Grid: 4 blocks x 512 threads (16 waves); each wave owns 2 N-tiles of 16.
// ---------------------------------------------------------------------------
__global__ __launch_bounds__(512) void jordan_scan_kernel(
    float* __restrict__ xin_h,          // [B*T, H] in: xin, out: h (in place)
    const float* __restrict__ last_logits,
    const bf16_t* __restrict__ Wy,  const float* __restrict__ by,
    const bf16_t* __restrict__ Wout, const float* __restrict__ bout,
    float* __restrict__ y_out,          // [B*T, O]
    float* __restrict__ y_last) {       // [B, O]
  __shared__ bf16_t sYp[16 * LDSTR];    // y_{t-1} slice, bf16 (A of GEMM1)
  __shared__ bf16_t sH [16 * LDSTR];    // h_t slice, bf16   (A of GEMM2)
  __shared__ float  sYf[16 * K_];       // y_{T-1} slice, f32 (for y_last)

  const int tid   = threadIdx.x;
  const int wave  = tid >> 5;
  const int lane  = tid & 31;
  const int half  = lane >> 4;
  const int l15   = lane & 15;
  const int bbase = blockIdx.x * 16;    // batch slice [bbase, bbase+16)

  // y_prev <- last_logits (f32 -> bf16)
  for (int i = tid; i < 16 * K_; i += blockDim.x) {
    int r = i >> 9, c = i & (K_ - 1);
    sYp[r * LDSTR + c] = (bf16_t)last_logits[(bbase + r) * K_ + c];
  }
  __syncthreads();

  const int n0 = wave * 32;             // two tiles: n0, n0+16

  for (int t = 0; t < T_; ++t) {
    // ---- phase 1: recur = y_prev @ W_y^T ; h = tanh(xin + recur + b_y) ----
    v8f acc[2] = {};
#pragma unroll
    for (int kk = 0; kk < K_; kk += 32) {
      const bf16_t* ap = &sYp[l15 * LDSTR + kk + 8 * half];
      v16bf a = make16(ap, ap + 16);
#pragma unroll
      for (int tI = 0; tI < 2; ++tI) {
        const bf16_t* bp = &Wy[(size_t)(n0 + tI * 16 + l15) * K_ + kk + 8 * half];
        acc[tI] = wmma_bf16(a, make16(bp, bp + 16), acc[tI]);
      }
    }
#pragma unroll
    for (int tI = 0; tI < 2; ++tI) {
      const int col = n0 + tI * 16 + l15;
      const float bias = by[col];
#pragma unroll
      for (int r = 0; r < 8; ++r) {
        const int bl = half * 8 + r;
        const size_t g = ((size_t)(bbase + bl) * T_ + t) * K_ + col;
        const float hv = tanhf(xin_h[g] + acc[tI][r] + bias);
        xin_h[g] = hv;                      // h output (in place over xin)
        sH[bl * LDSTR + col] = (bf16_t)hv;  // A operand for phase 2
      }
    }
    __syncthreads();

    // ---- phase 2: y = h @ W_out^T + b_out ----
    v8f acy[2] = {};
#pragma unroll
    for (int kk = 0; kk < K_; kk += 32) {
      const bf16_t* ap = &sH[l15 * LDSTR + kk + 8 * half];
      v16bf a = make16(ap, ap + 16);
#pragma unroll
      for (int tI = 0; tI < 2; ++tI) {
        const bf16_t* bp = &Wout[(size_t)(n0 + tI * 16 + l15) * K_ + kk + 8 * half];
        acy[tI] = wmma_bf16(a, make16(bp, bp + 16), acy[tI]);
      }
    }
#pragma unroll
    for (int tI = 0; tI < 2; ++tI) {
      const int col = n0 + tI * 16 + l15;
      const float bias = bout[col];
#pragma unroll
      for (int r = 0; r < 8; ++r) {
        const int bl = half * 8 + r;
        const size_t g = ((size_t)(bbase + bl) * T_ + t) * K_ + col;
        const float yv = acy[tI][r] + bias;
        y_out[g] = yv;
        sYp[bl * LDSTR + col] = (bf16_t)yv;   // becomes y_prev for t+1
        if (t == T_ - 1) sYf[bl * K_ + col] = yv;
      }
    }
    __syncthreads();
  }

  // y_last <- y_{T-1} (from LDS, avoids cross-wave global visibility issues)
  for (int i = tid; i < 16 * K_; i += blockDim.x) {
    int r = i >> 9, c = i & (K_ - 1);
    y_last[(bbase + r) * K_ + c] = sYf[r * K_ + c];
  }
}

// ---------------------------------------------------------------------------
// Host-side launcher.
// Inputs: 0=embeddings[B,T,I] 1=last_logits[B,O] 2=W_in[H,I] 3=b_in[H]
//         4=W_y[H,O] 5=b_y[H] 6=W_out[O,H] 7=b_out[O]   (all f32)
// d_out = h[B,T,H] | y[B,T,O] | y_last[B,O]  (f32, concatenated)
// d_ws  = bf16 copies of W_in, W_y, W_out (3 x 512 KB)
// ---------------------------------------------------------------------------
extern "C" void kernel_launch(void* const* d_in, const int* in_sizes, int n_in,
                              void* d_out, int out_size, void* d_ws, size_t ws_size,
                              hipStream_t stream) {
  const float* emb    = (const float*)d_in[0];
  const float* ylast0 = (const float*)d_in[1];
  const float* W_in   = (const float*)d_in[2];
  const float* b_in   = (const float*)d_in[3];
  const float* W_y    = (const float*)d_in[4];
  const float* b_y    = (const float*)d_in[5];
  const float* W_out  = (const float*)d_in[6];
  const float* b_out  = (const float*)d_in[7];

  const int WN = K_ * K_;               // 262144 elements per weight matrix
  bf16_t* wb_in  = (bf16_t*)d_ws;
  bf16_t* wb_y   = wb_in + WN;
  bf16_t* wb_out = wb_y + WN;

  float* out      = (float*)d_out;
  const size_t HN = (size_t)B_ * T_ * K_;   // 16,777,216
  float* h_region = out;                    // holds xin, then h
  float* y_region = out + HN;
  float* ylast    = out + 2 * HN;

  // 0) weights -> bf16 (every call: deterministic, inputs never mutated)
  cvt_bf16_kernel<<<(WN + 255) / 256, 256, 0, stream>>>(W_in,  wb_in,  WN);
  cvt_bf16_kernel<<<(WN + 255) / 256, 256, 0, stream>>>(W_y,   wb_y,   WN);
  cvt_bf16_kernel<<<(WN + 255) / 256, 256, 0, stream>>>(W_out, wb_out, WN);

  // 1) xin = emb @ W_in^T + b_in  (written into the h output region)
  xin_gemm_kernel<<<(B_ * T_) / 32, 256, 0, stream>>>(emb, wb_in, b_in, h_region);

  // 2) sequential scan: 4 persistent WGs, one 16-row batch slice each
  jordan_scan_kernel<<<B_ / 16, 512, 0, stream>>>(
      h_region, ylast0, wb_y, b_y, wb_out, b_out, y_region, ylast);
}
